// SelectiveScan_36490042146952
// MI455X (gfx1250) — matrix operand order repaired and checked by reading, not compiled, not run on hardware
//
#include <hip/hip_runtime.h>

typedef __bf16 bf16;
typedef __attribute__((ext_vector_type(16))) __bf16 v16bf;
typedef __attribute__((ext_vector_type(8)))  __bf16 v8bf;
typedef __attribute__((ext_vector_type(8)))  float  v8f;

#define D_MODEL 1024
#define D_STATE 16
#define D_INNER 2048
#define BATCH   8
#define SEQ     2048
#define M_TOTAL (BATCH*SEQ)   // 16384 rows

// ---------------- fp32 -> bf16 conversion ----------------
__global__ void f32_to_bf16_kernel(const float* __restrict__ in, bf16* __restrict__ out, int n) {
    int i = blockIdx.x * blockDim.x + threadIdx.x;
    if (i < n) out[i] = (bf16)in[i];
}

// ---------------- WMMA GEMM: C[M,N] = A[M,K] @ W[N,K]^T ----------------
// A, W bf16 row-major; C fp32 row-major (ldc = N).
// Block = 256 threads = 8 waves arranged 4(M) x 2(N); each wave computes a
// 16(M) x 64(N) tile (4 accumulators sharing one A fragment per K-step).
// Block tile = 64(M) x 128(N). M % 64 == 0, N % 128 == 0, K % 32 == 0 here.
__global__ void wmma_gemm_bf16(const bf16* __restrict__ A, const bf16* __restrict__ W,
                               float* __restrict__ C, int K, int N) {
    const int lane = threadIdx.x & 31;
    const int wid  = threadIdx.x >> 5;
    const int m0   = blockIdx.x * 64  + (wid & 3) * 16;
    const int n0   = blockIdx.y * 128 + (wid >> 2) * 64;
    const int r    = lane & 15;
    const int half = lane >> 4;

    // A fragment (16x32 bf16): lane holds row m0+r; elems 0-7 = K[half*16 .. +7],
    // elems 8-15 = K[half*16+8 .. +7]  (contiguous k order, matches B)
    const bf16* Aptr = A + (size_t)(m0 + r) * K + half * 16;
    // B fragment (32x16 bf16): lane holds col n+r; elems j = W[n+r, k0+half*16+j]
    const bf16* Bptr = W + (size_t)(n0 + r) * K + half * 16;
    const size_t bstep = (size_t)16 * K;   // next 16-column group of W

    v8f acc0 = {}, acc1 = {}, acc2 = {}, acc3 = {};
    for (int k0 = 0; k0 < K; k0 += 32) {
        v8bf alo = *(const v8bf*)(Aptr + k0);
        v8bf ahi = *(const v8bf*)(Aptr + k0 + 8);
        v16bf afrag = __builtin_shufflevector(alo, ahi,
                        0,1,2,3,4,5,6,7,8,9,10,11,12,13,14,15);
        v16bf b0 = *(const v16bf*)(Bptr + k0);
        v16bf b1 = *(const v16bf*)(Bptr + bstep     + k0);
        v16bf b2 = *(const v16bf*)(Bptr + bstep * 2 + k0);
        v16bf b3 = *(const v16bf*)(Bptr + bstep * 3 + k0);
        __builtin_prefetch(Aptr + k0 + 256, 0, 1);
        __builtin_prefetch(Bptr + k0 + 256, 0, 1);
        acc0 = __builtin_amdgcn_wmma_f32_16x16x32_bf16(
                   false, afrag, false, b0, (short)0, acc0, false, false);
        acc1 = __builtin_amdgcn_wmma_f32_16x16x32_bf16(
                   false, afrag, false, b1, (short)0, acc1, false, false);
        acc2 = __builtin_amdgcn_wmma_f32_16x16x32_bf16(
                   false, afrag, false, b2, (short)0, acc2, false, false);
        acc3 = __builtin_amdgcn_wmma_f32_16x16x32_bf16(
                   false, afrag, false, b3, (short)0, acc3, false, false);
    }
    // D layout: VGPR v -> row m0 + 8*half + v, col n + (lane&15)
    float* Crow = C + (size_t)(m0 + 8 * half) * N + n0 + r;
    #pragma unroll
    for (int v = 0; v < 8; ++v) {
        Crow[(size_t)v * N]      = acc0[v];
        Crow[(size_t)v * N + 16] = acc1[v];
        Crow[(size_t)v * N + 32] = acc2[v];
        Crow[(size_t)v * N + 48] = acc3[v];
    }
}

// ---------------- depthwise causal conv (k=4) + SiLU ----------------
// x_inner is the first D_INNER columns of xz (ld = 2*D_INNER)
__global__ void conv_silu_kernel(const float* __restrict__ xz, const float* __restrict__ cw,
                                 const float* __restrict__ cb, float* __restrict__ xconv) {
    int i = blockIdx.x * blockDim.x + threadIdx.x;    // over M_TOTAL * D_INNER
    if (i >= M_TOTAL * D_INNER) return;
    int d  = i & (D_INNER - 1);
    int ml = i >> 11;                 // ml = b*SEQ + l
    int l  = ml & (SEQ - 1);
    float acc = cb[d];
    const float* w = cw + d * 4;
    #pragma unroll
    for (int j = 0; j < 4; ++j) {
        int lj = l - 3 + j;
        if (lj >= 0)
            acc += w[j] * xz[(size_t)(ml - 3 + j) * (2 * D_INNER) + d];
    }
    acc = acc / (1.f + __expf(-acc));  // SiLU
    xconv[i] = acc;
}

// ---------------- xp = x_conv @ W_x^T   (N = 33, VALU + wave reduce) ----------------
__global__ void xp_kernel(const float* __restrict__ xconv, const float* __restrict__ Wx,
                          float* __restrict__ xp) {
    __shared__ float s[33];
    const int m   = blockIdx.x;
    const int tid = threadIdx.x;
    if (tid < 33) s[tid] = 0.f;
    __syncthreads();
    float acc[33];
    #pragma unroll
    for (int n = 0; n < 33; ++n) acc[n] = 0.f;
    const float* xr = xconv + (size_t)m * D_INNER;
    for (int k = tid; k < D_INNER; k += 256) {
        float xv = xr[k];
        #pragma unroll
        for (int n = 0; n < 33; ++n) acc[n] += xv * Wx[n * D_INNER + k];
    }
    #pragma unroll
    for (int n = 0; n < 33; ++n) {
        float v = acc[n];
        v += __shfl_down(v, 16, 32);
        v += __shfl_down(v, 8, 32);
        v += __shfl_down(v, 4, 32);
        v += __shfl_down(v, 2, 32);
        v += __shfl_down(v, 1, 32);
        if ((tid & 31) == 0) atomicAdd(&s[n], v);
    }
    __syncthreads();
    if (tid < 33) xp[(size_t)m * 33 + tid] = s[tid];
}

// ---------------- selective scan, fused with D-skip and SiLU(z) gate ----------------
// 16 lanes handle one (b, d) pair: lane s owns state s. y_t = sum_s h[s]*C[s]
// reduced with shfl_xor within the aligned 16-lane group (wave32).
__global__ void scan_kernel(const float* __restrict__ xp, const float* __restrict__ xconv,
                            const float* __restrict__ xz, const float* __restrict__ Wdt,
                            const float* __restrict__ bdt, const float* __restrict__ Alog,
                            const float* __restrict__ Dp, bf16* __restrict__ ybf) {
    const int tid = threadIdx.x;
    const int g   = blockIdx.x * (256 / 16) + (tid >> 4);  // (b,d) pair id
    const int s   = tid & 15;
    const int b   = g >> 11;            // g / D_INNER
    const int d   = g & (D_INNER - 1);

    const float A_s   = -__expf(Alog[d * D_STATE + s]);
    const float wdt   = Wdt[d];
    const float bdt_v = bdt[d];
    const float Dv    = Dp[d];

    float h = 0.f;
    const size_t xpb = (size_t)b * SEQ * 33;
    const size_t xcb = (size_t)b * SEQ * D_INNER + d;
    const size_t zb  = (size_t)b * SEQ * (2 * D_INNER) + D_INNER + d;

    for (int l = 0; l < SEQ; ++l) {
        const float* xpl = xp + xpb + (size_t)l * 33;
        float Bt = xpl[s];
        float Ct = xpl[16 + s];
        float xl = xpl[32];
        float pre = xl * wdt + bdt_v;
        float dt  = (pre > 20.f) ? pre : log1pf(__expf(pre));   // softplus
        float xc  = xconv[xcb + (size_t)l * D_INNER];
        h = __expf(A_s * dt) * h + (Bt * dt) * xc;
        float v = h * Ct;
        v += __shfl_xor(v, 8, 16);
        v += __shfl_xor(v, 4, 16);
        v += __shfl_xor(v, 2, 16);
        v += __shfl_xor(v, 1, 16);
        if (s == 0) {
            float y = v + xc * Dv;
            float z = xz[zb + (size_t)l * (2 * D_INNER)];
            y *= z / (1.f + __expf(-z));                        // * SiLU(z)
            ybf[xcb + (size_t)l * D_INNER] = (bf16)y;
        }
    }
}

// ---------------- host launch ----------------
extern "C" void kernel_launch(void* const* d_in, const int* in_sizes, int n_in,
                              void* d_out, int out_size, void* d_ws, size_t ws_size,
                              hipStream_t stream) {
    const float* x      = (const float*)d_in[0];
    const float* W_in   = (const float*)d_in[1];
    const float* conv_w = (const float*)d_in[2];
    const float* conv_b = (const float*)d_in[3];
    const float* W_x    = (const float*)d_in[4];
    const float* W_dt   = (const float*)d_in[5];
    const float* b_dt   = (const float*)d_in[6];
    const float* A_log  = (const float*)d_in[7];
    const float* D_par  = (const float*)d_in[8];
    const float* W_out  = (const float*)d_in[9];
    float* out = (float*)d_out;

    char* ws = (char*)d_ws;
    size_t off = 0;
    auto alloc = [&](size_t bytes) -> void* {
        off = (off + 255) & ~(size_t)255;
        void* p = ws + off;
        off += bytes;
        return p;
    };
    bf16*  x_bf    = (bf16*) alloc((size_t)M_TOTAL * D_MODEL * 2);
    bf16*  Win_bf  = (bf16*) alloc((size_t)2 * D_INNER * D_MODEL * 2);
    bf16*  Wout_bf = (bf16*) alloc((size_t)D_MODEL * D_INNER * 2);
    float* xz      = (float*)alloc((size_t)M_TOTAL * 2 * D_INNER * 4);
    float* xconv   = (float*)alloc((size_t)M_TOTAL * D_INNER * 4);
    float* xp      = (float*)alloc((size_t)M_TOTAL * 33 * 4);
    bf16*  y_bf    = (bf16*) alloc((size_t)M_TOTAL * D_INNER * 2);
    (void)ws_size; (void)in_sizes; (void)n_in; (void)out_size;

    int n;
    n = M_TOTAL * D_MODEL;
    f32_to_bf16_kernel<<<(n + 255) / 256, 256, 0, stream>>>(x, x_bf, n);
    n = 2 * D_INNER * D_MODEL;
    f32_to_bf16_kernel<<<(n + 255) / 256, 256, 0, stream>>>(W_in, Win_bf, n);
    n = D_MODEL * D_INNER;
    f32_to_bf16_kernel<<<(n + 255) / 256, 256, 0, stream>>>(W_out, Wout_bf, n);

    // GEMM1: xz[16384, 4096] = x_bf[16384,1024] @ W_in[4096,1024]^T
    wmma_gemm_bf16<<<dim3(M_TOTAL / 64, (2 * D_INNER) / 128), 256, 0, stream>>>(
        x_bf, Win_bf, xz, D_MODEL, 2 * D_INNER);

    // depthwise conv + SiLU
    n = M_TOTAL * D_INNER;
    conv_silu_kernel<<<(n + 255) / 256, 256, 0, stream>>>(xz, conv_w, conv_b, xconv);

    // xp = x_conv @ W_x^T  (N = 33)
    xp_kernel<<<M_TOTAL, 256, 0, stream>>>(xconv, W_x, xp);

    // selective scan (+ D skip, SiLU(z) gate) -> y_bf
    scan_kernel<<<(M_TOTAL) / 16, 256, 0, stream>>>(
        xp, xconv, xz, W_dt, b_dt, A_log, D_par, y_bf);

    // GEMM2: out[16384, 1024] = y_bf[16384,2048] @ W_out[1024,2048]^T
    wmma_gemm_bf16<<<dim3(M_TOTAL / 64, D_MODEL / 128), 256, 0, stream>>>(
        y_bf, Wout_bf, out, D_INNER, D_MODEL);
}
